// Qwen3VLTextAttention_49649821941840
// MI455X (gfx1250) — matrix operand order repaired and checked by reading, not compiled, not run on hardware
//
#include <hip/hip_runtime.h>
#include <hip/hip_bf16.h>

// ---------------------------------------------------------------------------
// Qwen3-VL text attention for MI455X (gfx1250), bf16 WMMA pipeline.
// T=2048, D=2560, N=32 q heads, K=8 kv heads, H=128, G=4 (GQA).
// ---------------------------------------------------------------------------

#define T_TOK 2048
#define D_MODEL 2560
#define NQ 32
#define NKV 8
#define HD 128
#define QKV_COLS 6144            // NQ*HD + 2*NKV*HD
#define NH (NQ * HD)             // 4096
#define KH (NKV * HD)            // 1024

typedef __attribute__((ext_vector_type(16))) __bf16 v16bf;
typedef __attribute__((ext_vector_type(8)))  __bf16 v8bf;
typedef __attribute__((ext_vector_type(8)))  float  v8f;

// ---------------- WMMA fragment access (CDNA5 16-bit layouts) --------------
// A (16x32, MxK): lanes 0-15 -> M=lane, K in {kb..kb+7, kb+16..kb+23}, kb=0
//                 lanes 16-31 -> M=lane-16, kb=8
__device__ __forceinline__ const __bf16* a_lane_ptr(const __bf16* base,
                                                    int lda, int m0) {
  const int lane = threadIdx.x & 31;
  return base + (size_t)(m0 + (lane & 15)) * lda + ((lane < 16) ? 0 : 8);
}
__device__ __forceinline__ v16bf load_a_at(const __bf16* p) {
  v8bf lo = *(const v8bf*)(p);
  v8bf hi = *(const v8bf*)(p + 16);
  v16bf r;
#pragma unroll
  for (int i = 0; i < 8; ++i) { r[i] = lo[i]; r[8 + i] = hi[i]; }
  return r;
}

// B (32x16, KxN), stored K-contiguous per column: baseT[col*ldb + k].
// lanes 0-15 -> N=lane, K in {0..7, 16..23} ; lanes 16-31 -> N=lane-16, K in {8..15, 24..31}
__device__ __forceinline__ const __bf16* b_lane_ptr(const __bf16* baseT,
                                                    int ldb, int n0) {
  const int lane = threadIdx.x & 31;
  return baseT + (size_t)(n0 + (lane & 15)) * ldb + ((lane < 16) ? 0 : 8);
}
__device__ __forceinline__ v16bf load_b_at(const __bf16* p) {
  v8bf lo = *(const v8bf*)(p);
  v8bf hi = *(const v8bf*)(p + 16);
  v16bf r;
#pragma unroll
  for (int i = 0; i < 8; ++i) { r[i] = lo[i]; r[8 + i] = hi[i]; }
  return r;
}

#define WMMA_BF16(a, b, c) \
  __builtin_amdgcn_wmma_f32_16x16x32_bf16(false, (a), false, (b), (short)0, (c), false, false)

// ---------------- Pack / convert kernels -----------------------------------
__global__ void cvt_f32_bf16(const float* __restrict__ src,
                             __bf16* __restrict__ dst, int n) {
  for (int i = blockIdx.x * blockDim.x + threadIdx.x; i < n;
       i += gridDim.x * blockDim.x)
    dst[i] = (__bf16)src[i];
}

// Fused QKV weight, stored transposed: wT[c * D + d], c in [0,6144)
__global__ void pack_wqkv(const float* __restrict__ qw,
                          const float* __restrict__ kw,
                          const float* __restrict__ vw,
                          __bf16* __restrict__ wT) {
  const int n = QKV_COLS * D_MODEL;
  for (int i = blockIdx.x * blockDim.x + threadIdx.x; i < n;
       i += gridDim.x * blockDim.x) {
    int c = i / D_MODEL, d = i - c * D_MODEL;
    float v;
    if (c < NH)            v = qw[(size_t)d * NH + c];
    else if (c < NH + KH)  v = kw[(size_t)d * KH + (c - NH)];
    else                   v = vw[(size_t)d * KH + (c - NH - KH)];
    wT[i] = (__bf16)v;
  }
}

// o_w (N,H,D) -> transposed oT[d * 4096 + c], c = n*H+h
__global__ void pack_ow(const float* __restrict__ ow, __bf16* __restrict__ oT) {
  const int n = D_MODEL * NH;
  for (int i = blockIdx.x * blockDim.x + threadIdx.x; i < n;
       i += gridDim.x * blockDim.x) {
    int d = i / NH, c = i - d * NH;
    oT[i] = (__bf16)ow[(size_t)c * D_MODEL + d];
  }
}

// ---------------- bf16 WMMA GEMM: C(MxN,f32) = A(MxK) * B(KxN) -------------
// B passed transposed (K-contiguous per column). One wave -> 32x64 tile,
// fragments double-buffered so loads for k+32 overlap WMMAs for k.
// launch_bounds(256,1): allow the ~200-VGPR tile without scratch spills.
__global__ void __launch_bounds__(256, 1)
gemm_bf16(const __bf16* __restrict__ A, const __bf16* __restrict__ BT,
          float* __restrict__ C, int M, int N, int Kd) {
  const int wid = (blockIdx.x * blockDim.x + threadIdx.x) >> 5;
  const int ntiles = N >> 6;
  const int mt = wid / ntiles, nt = wid - mt * ntiles;
  if (mt * 32 >= M) return;
  const int m0 = mt * 32, n0 = nt * 64;

  // lane-resolved base pointers; k-offsets fold into load immediates
  const __bf16* pa0 = a_lane_ptr(A, Kd, m0);
  const __bf16* pa1 = a_lane_ptr(A, Kd, m0 + 16);
  const __bf16* pb0 = b_lane_ptr(BT, Kd, n0);
  const __bf16* pb1 = b_lane_ptr(BT, Kd, n0 + 16);
  const __bf16* pb2 = b_lane_ptr(BT, Kd, n0 + 32);
  const __bf16* pb3 = b_lane_ptr(BT, Kd, n0 + 48);

  v8f acc[2][4] = {};
  v16bf af[2][2], bf[2][4];
  af[0][0] = load_a_at(pa0); af[0][1] = load_a_at(pa1);
  bf[0][0] = load_b_at(pb0); bf[0][1] = load_b_at(pb1);
  bf[0][2] = load_b_at(pb2); bf[0][3] = load_b_at(pb3);

  int cur = 0;
  // steady state: prefetch k0+32 while computing k0
  for (int k0 = 0; k0 < Kd - 32; k0 += 32) {
    const int nxt = cur ^ 1;
    af[nxt][0] = load_a_at(pa0 + k0 + 32);
    af[nxt][1] = load_a_at(pa1 + k0 + 32);
    bf[nxt][0] = load_b_at(pb0 + k0 + 32);
    bf[nxt][1] = load_b_at(pb1 + k0 + 32);
    bf[nxt][2] = load_b_at(pb2 + k0 + 32);
    bf[nxt][3] = load_b_at(pb3 + k0 + 32);
#pragma unroll
    for (int i = 0; i < 2; ++i)
#pragma unroll
      for (int j = 0; j < 4; ++j)
        acc[i][j] = WMMA_BF16(af[cur][i], bf[cur][j], acc[i][j]);
    cur = nxt;
  }
  // final k-step
#pragma unroll
  for (int i = 0; i < 2; ++i)
#pragma unroll
    for (int j = 0; j < 4; ++j)
      acc[i][j] = WMMA_BF16(af[cur][i], bf[cur][j], acc[i][j]);

  const int lane = threadIdx.x & 31;
  const int c16 = lane & 15;
  const int rb = (lane < 16) ? 0 : 8;
#pragma unroll
  for (int i = 0; i < 2; ++i)
#pragma unroll
    for (int j = 0; j < 4; ++j)
#pragma unroll
      for (int r = 0; r < 8; ++r)
        C[(size_t)(m0 + 16 * i + rb + r) * N + (n0 + j * 16 + c16)] =
            acc[i][j][r];
}

// ---------------- RMSNorm + mRoPE (+ v transpose) --------------------------
// One wave per (token, head-slot). slots: 0..31 q, 32..39 k, 40..47 v.
__global__ void normrope(const float* __restrict__ qkv,
                         const int* __restrict__ pos,
                         const float* __restrict__ qnw,
                         const float* __restrict__ knw,
                         __bf16* __restrict__ qb,   // [n][t][h]
                         __bf16* __restrict__ kb,   // [kh][t][h]
                         __bf16* __restrict__ vb) { // [kh][h][t]
  const int gw = (blockIdx.x * blockDim.x + threadIdx.x) >> 5;
  const int lane = threadIdx.x & 31;
  const int t = gw / 48, hd = gw - t * 48;
  if (t >= T_TOK) return;

  if (hd >= 40) {  // v: just convert + transpose
    const int kh = hd - 40;
    const float* x = qkv + (size_t)t * QKV_COLS + NH + KH + kh * HD;
#pragma unroll
    for (int e = 0; e < 4; ++e) {
      int h = lane + 32 * e;
      vb[((size_t)kh * HD + h) * T_TOK + t] = (__bf16)x[h];
    }
    return;
  }

  const bool is_q = (hd < 32);
  const float* w = is_q ? qnw : knw;
  const float* x = qkv + (size_t)t * QKV_COLS +
                   (is_q ? hd * HD : NH + (hd - 32) * HD);
  float xv[4], ss = 0.f;
#pragma unroll
  for (int e = 0; e < 4; ++e) {
    xv[e] = x[lane + 32 * e];
    ss += xv[e] * xv[e];
  }
#pragma unroll
  for (int off = 1; off < 32; off <<= 1) ss += __shfl_xor(ss, off, 32);
  const float rstd = rsqrtf(ss * (1.0f / HD) + 1e-6f);
#pragma unroll
  for (int e = 0; e < 4; ++e) xv[e] = xv[e] * rstd * w[lane + 32 * e];

  __bf16* dst = is_q ? (qb + ((size_t)hd * T_TOK + t) * HD)
                     : (kb + ((size_t)(hd - 32) * T_TOK + t) * HD);
  // pairs (j, j+64): j = lane, lane+32  (xv[0],xv[2]) and (xv[1],xv[3])
#pragma unroll
  for (int pi = 0; pi < 2; ++pi) {
    const int j = lane + 32 * pi;
    const int m3 = j % 3;
    const int sel = (j >= 60) ? 0 : m3;  // mrope sections (24,20,20)
    const float p = (float)pos[sel * T_TOK + t];
    // inv_freq[j] = theta^{-(2j/128)}, ln(5e6) = 15.4249484703984
    const float freq = p * __expf(-(float)j * (15.424948470398375f / 64.0f));
    const float c = __cosf(freq), s = __sinf(freq);
    const float x1 = xv[pi], x2 = xv[pi + 2];
    dst[j]      = (__bf16)(x1 * c - x2 * s);
    dst[j + 64] = (__bf16)(x2 * c + x1 * s);
  }
}

// ---------------- Flash attention (causal, GQA) ----------------------------
// One wave per (head, 16-row q tile). Keys processed in blocks of 32.
// V fragments are issued before the softmax so their latency hides under
// the exp/shuffle VALU work. launch_bounds(256,1): ~280 live VGPRs, no spill.
__global__ void __launch_bounds__(256, 1)
flash_attn(const __bf16* __restrict__ qb, const __bf16* __restrict__ kb,
           const __bf16* __restrict__ vb,
           __bf16* __restrict__ ctxb) { // [t][n*H+h]
  __shared__ __bf16 lds_p[8][16 * 32];  // per-wave P staging tile
  const int widx = threadIdx.x >> 5;
  const int lane = threadIdx.x & 31;
  const int gw = blockIdx.x * 8 + widx;
  const int head = gw >> 7;          // 128 q tiles per head
  const int qblk = gw & 127;
  const int q0 = qblk * 16;
  const int kh = head >> 2;          // GQA: 4 q heads per kv head
  const int hl = (lane >= 16) ? 1 : 0;
  const int c16 = lane & 15;
  const float scale = 0.08838834764831845f;  // 1/sqrt(128)

  const __bf16* qbase = qb + (size_t)head * T_TOK * HD;
  const __bf16* kbase = kb + (size_t)kh * T_TOK * HD;
  const __bf16* vbase = vb + (size_t)kh * HD * T_TOK;

  // q fragments for the whole row of 4 K-steps over H
  v16bf qf[4];
  {
    const __bf16* pq = a_lane_ptr(qbase, HD, q0);
#pragma unroll
    for (int kk = 0; kk < 4; ++kk) qf[kk] = load_a_at(pq + kk * 32);
  }
  // lane-resolved base pointers, advanced by one 32-key block per iteration
  const __bf16* pk = b_lane_ptr(kbase, HD, 0);      // +32*HD per block
  const __bf16* pv = b_lane_ptr(vbase, T_TOK, 0);   // +32 per block

  v8f O[8] = {};
  float rowm[8], rowl[8];
#pragma unroll
  for (int r = 0; r < 8; ++r) { rowm[r] = -1e30f; rowl[r] = 0.f; }

  for (int k0 = 0; k0 <= q0 + 15; k0 += 32, pk += 32 * HD, pv += 32) {
    // ---- K fragments for 32 keys (one load clause), then S = q @ k^T
    v16bf kf[2][4];
#pragma unroll
    for (int sub = 0; sub < 2; ++sub)
#pragma unroll
      for (int kk = 0; kk < 4; ++kk)
        kf[sub][kk] = load_b_at(pk + sub * 16 * HD + kk * 32);
    v8f S0 = {}, S1 = {};
#pragma unroll
    for (int kk = 0; kk < 4; ++kk) {
      S0 = WMMA_BF16(qf[kk], kf[0][kk], S0);
      S1 = WMMA_BF16(qf[kk], kf[1][kk], S1);
    }
    // ---- issue V fragment loads early (independent of softmax)
    v16bf vf[8];
#pragma unroll
    for (int j = 0; j < 8; ++j) vf[j] = load_b_at(pv + j * 16 * T_TOK);
    // ---- online softmax on C-layout fragments (row = r + 8*hl)
#pragma unroll
    for (int r = 0; r < 8; ++r) {
      const int qt = q0 + r + 8 * hl;
      float s0 = S0[r] * scale; if (k0 + c16 > qt)      s0 = -1e30f;
      float s1 = S1[r] * scale; if (k0 + 16 + c16 > qt) s1 = -1e30f;
      float m = fmaxf(s0, s1);
#pragma unroll
      for (int off = 1; off < 16; off <<= 1) m = fmaxf(m, __shfl_xor(m, off, 32));
      const float mnew = fmaxf(rowm[r], m);
      const float corr = __expf(rowm[r] - mnew);
      const float p0 = __expf(s0 - mnew);
      const float p1 = __expf(s1 - mnew);
      float rs = p0 + p1;
#pragma unroll
      for (int off = 1; off < 16; off <<= 1) rs += __shfl_xor(rs, off, 32);
      rowl[r] = rowl[r] * corr + rs;
      rowm[r] = mnew;
#pragma unroll
      for (int j = 0; j < 8; ++j) O[j][r] *= corr;
      lds_p[widx][(r + 8 * hl) * 32 + c16]      = (__bf16)p0;
      lds_p[widx][(r + 8 * hl) * 32 + 16 + c16] = (__bf16)p1;
    }
    // wave-private LDS round trip: C-layout -> A-layout
    asm volatile("s_wait_dscnt 0x0" ::: "memory");
    v16bf pf;
    {
      const __bf16* p = &lds_p[widx][c16 * 32 + (hl ? 8 : 0)];
#pragma unroll
      for (int i = 0; i < 8; ++i) { pf[i] = p[i]; pf[8 + i] = p[16 + i]; }
    }
    // ---- O += P @ V  (V stored [kh][h][t]: K-contiguous columns)
#pragma unroll
    for (int j = 0; j < 8; ++j) O[j] = WMMA_BF16(pf, vf[j], O[j]);
  }

  // ---- epilogue: normalize and emit ctx as bf16 [t][n*H+h]
#pragma unroll
  for (int j = 0; j < 8; ++j)
#pragma unroll
    for (int r = 0; r < 8; ++r) {
      const int t = q0 + r + 8 * hl;
      ctxb[(size_t)t * NH + head * HD + j * 16 + c16] =
          (__bf16)(O[j][r] / rowl[r]);
    }
}

// ---------------------------------------------------------------------------
extern "C" void kernel_launch(void* const* d_in, const int* in_sizes, int n_in,
                              void* d_out, int out_size, void* d_ws,
                              size_t ws_size, hipStream_t stream) {
  const float* hs  = (const float*)d_in[0];
  const int*   pos = (const int*)d_in[1];
  const float* qw  = (const float*)d_in[2];
  const float* kw  = (const float*)d_in[3];
  const float* vw  = (const float*)d_in[4];
  const float* ow  = (const float*)d_in[5];
  const float* qnw = (const float*)d_in[6];
  const float* knw = (const float*)d_in[7];
  float* out = (float*)d_out;

  char* ws = (char*)d_ws;
  constexpr size_t SZ_HSB  = (size_t)T_TOK * D_MODEL * 2;      // 10.5 MB
  constexpr size_t SZ_WQKV = (size_t)QKV_COLS * D_MODEL * 2;   // 31.5 MB
  constexpr size_t SZ_OWT  = (size_t)D_MODEL * NH * 2;         // 21.0 MB
  constexpr size_t SZ_QKV  = (size_t)T_TOK * QKV_COLS * 4;     // 50.3 MB
  constexpr size_t SZ_QB   = (size_t)NQ * T_TOK * HD * 2;      // 16.8 MB
  constexpr size_t SZ_KB   = (size_t)NKV * T_TOK * HD * 2;     //  4.2 MB
  constexpr size_t SZ_VB   = (size_t)NKV * HD * T_TOK * 2;     //  4.2 MB
  size_t o = 0;
  __bf16* hsb   = (__bf16*)(ws + o); o += SZ_HSB;
  __bf16* wqkvT = (__bf16*)(ws + o); o += SZ_WQKV;
  __bf16* owT   = (__bf16*)(ws + o); o += SZ_OWT;
  float*  qkv   = (float*)(ws + o);  o += SZ_QKV;
  __bf16* qb    = (__bf16*)(ws + o); o += SZ_QB;
  __bf16* kb    = (__bf16*)(ws + o); o += SZ_KB;
  __bf16* vb    = (__bf16*)(ws + o); o += SZ_VB;
  __bf16* ctxb  = (__bf16*)(ws + o);  // T*NH*2 = 16.8 MB  (total ~155 MB)

  // 1) bf16 packing (one-time; all data then L2-resident)
  cvt_f32_bf16<<<2048, 256, 0, stream>>>(hs, hsb, T_TOK * D_MODEL);
  pack_wqkv<<<4096, 256, 0, stream>>>(qw, kw, vw, wqkvT);
  pack_ow<<<4096, 256, 0, stream>>>(ow, owT);

  // 2) fused QKV GEMM: (2048x2560) x (2560x6144) -> f32
  {
    int waves = (T_TOK / 32) * (QKV_COLS / 64);  // 6144
    gemm_bf16<<<waves / 8, 256, 0, stream>>>(hsb, wqkvT, qkv,
                                             T_TOK, QKV_COLS, D_MODEL);
  }

  // 3) RMSNorm + mRoPE + v transpose
  normrope<<<(T_TOK * 48) / 8, 256, 0, stream>>>(qkv, pos, qnw, knw, qb, kb, vb);

  // 4) causal flash attention (GQA 4:1), ctx in bf16
  flash_attn<<<(NQ * (T_TOK / 16)) / 8, 256, 0, stream>>>(qb, kb, vb, ctxb);

  // 5) output projection: (2048x4096) x (4096x2560) -> f32 out
  {
    int waves = (T_TOK / 32) * (D_MODEL / 64);  // 2560
    gemm_bf16<<<waves / 8, 256, 0, stream>>>(ctxb, owT, out,
                                             T_TOK, D_MODEL, NH);
  }
}
